// AttentionalCopula_87668872446460
// MI455X (gfx1250) — matrix operand order repaired and checked by reading, not compiled, not run on hardware
//
#include <hip/hip_runtime.h>
#include <stdint.h>

// ---------------------------------------------------------------------------
// AttentionalCopula forward for MI455X (gfx1250, wave32, WMMA bf16 path)
// ---------------------------------------------------------------------------

#define DEV __device__ __forceinline__

typedef __attribute__((ext_vector_type(16))) __bf16 v16bf;
typedef __attribute__((ext_vector_type(8)))  __bf16 v8bf;
typedef __attribute__((ext_vector_type(8)))  float  v8f;

#define B_  8
#define V_  1024
#define D_  256
#define H_  8
#define A_  64
#define M_  128
#define P_  256
#define W_  128
#define HA_ 512
#define R_  10

// ---- WMMA fragment helpers (CDNA5 ISA 7.12.2 layouts, wave32) -------------
// A (16x32 bf16): lane half h2: K = h2*8 + {0..7} then +16 + {0..7}
DEV v16bf frag_a(const __bf16* p) {
  v8bf lo = *(const v8bf*)(p);
  v8bf hi = *(const v8bf*)(p + 16);
  v16bf r;
#pragma unroll
  for (int i = 0; i < 8; ++i) { r[i] = lo[i]; r[i + 8] = hi[i]; }
  return r;
}
// B (32x16 bf16, fed from W^T rows): lane half h2: K = h2*8 + {0..7} then +16 + {0..7}
DEV v16bf frag_b(const __bf16* p) {
  v8bf lo = *(const v8bf*)(p);
  v8bf hi = *(const v8bf*)(p + 16);
  v16bf r;
#pragma unroll
  for (int i = 0; i < 8; ++i) { r[i] = lo[i]; r[i + 8] = hi[i]; }
  return r;
}
DEV v8f wmma_bf16(v16bf a, v16bf b, v8f c) {
  return __builtin_amdgcn_wmma_f32_16x16x32_bf16(false, a, false, b, (short)0, c,
                                                 false, false);
}

// ---------------------------------------------------------------------------
// Prep kernels
// ---------------------------------------------------------------------------
__global__ void zero_kernel(float* p, int n) {
  int i = blockIdx.x * blockDim.x + threadIdx.x;
  if (i < n) p[i] = 0.f;
}

__global__ void f2b_kernel(const float* __restrict__ s, __bf16* __restrict__ d, long n) {
  for (long i = (long)blockIdx.x * blockDim.x + threadIdx.x; i < n;
       i += (long)gridDim.x * blockDim.x)
    d[i] = (__bf16)s[i];
}

// src[b][i][o] (f32) -> dst[b][o][i<dinKeep] (bf16)
__global__ void convT_kernel(const float* __restrict__ src, __bf16* __restrict__ dst,
                             int batch, int din, int dout, int dinKeep,
                             long sStride, long dStride) {
  long total = (long)batch * din * dout;
  for (long x = (long)blockIdx.x * blockDim.x + threadIdx.x; x < total;
       x += (long)gridDim.x * blockDim.x) {
    long bi  = x / ((long)din * dout);
    long rem = x - bi * (long)din * dout;
    int  i   = (int)(rem / dout);
    int  oc  = (int)(rem - (long)i * dout);
    if (i < dinKeep)
      dst[bi * dStride + (long)oc * dinKeep + i] = (__bf16)src[bi * sStride + rem];
  }
}

// dst[b*P+p][0..255] = kvx[b][perm[p]][0..255]
__global__ void gather_rows_kernel(const __bf16* __restrict__ src, const int* __restrict__ perm,
                                   __bf16* __restrict__ dst) {
  int row = blockIdx.x;              // B_*P_
  int b = row >> 8, p = row & 255;
  const __bf16* s = src + ((size_t)b * V_ + (size_t)perm[p]) * D_;
  __bf16* d = dst + (size_t)row * D_;
  for (int i = threadIdx.x; i < D_; i += blockDim.x) d[i] = s[i];
}

// ---------------------------------------------------------------------------
// Fused key/value head-MLP:   X(64x256) -> relu(.W0+b0+u*w0u) -> relu(.W1+b1)
//                             -> .W2+b2   all via v_wmma_f32_16x16x32_bf16
// grid = 4096 blocks: (l, t, b, h, vtile16)
// ---------------------------------------------------------------------------
struct KvBias {
  const float* W0[4];  // original (8,257,128) f32: row 256 = u weights
  const float* b0[4];
  const float* b1[4];
  const float* b2[4];
};

__global__ __launch_bounds__(256) void kv_mlp_kernel(
    const __bf16* __restrict__ kvx,   // [B][V][256] bf16
    const float*  __restrict__ trueu, // [B][V]
    const __bf16* __restrict__ W0T,   // [lt][h][128][256]
    const __bf16* __restrict__ W1T,   // [lt][h][128][128]
    const __bf16* __restrict__ W2T,   // [lt][h][64][128]
    KvBias kb,
    __bf16* __restrict__ kout,        // [l][b][h][V][64]
    __bf16* __restrict__ vout) {
  __shared__ __align__(16) __bf16 Xs[64 * 256];
  __shared__ __align__(16) __bf16 Ys[64 * 128];
  __shared__ float Us[64];

  int id = blockIdx.x;
  int vt = id & 15;
  int h  = (id >> 4) & 7;
  int b  = (id >> 7) & 7;
  int t  = (id >> 10) & 1;
  int l  = (id >> 11) & 1;
  int lt = l * 2 + t;

  int tid = threadIdx.x, wave = tid >> 5, lane = tid & 31;
  int m16 = lane & 15, h2 = lane >> 4;
  int v0 = vt * 64;

  // --- stage activations ---------------------------------------------------
  const __bf16* xsrc = kvx + ((size_t)b * V_ + v0) * D_;
  for (int i = tid; i < 64 * 256 / 8; i += 256)
    ((v8bf*)Xs)[i] = ((const v8bf*)xsrc)[i];
  if (tid < 64) Us[tid] = trueu[(size_t)b * V_ + v0 + tid];
  __syncthreads();

  int tmW = wave & 3;

  // --- stage 1: (64x256)@(256x128) + b0 + u*w0u, relu ----------------------
  {
    const __bf16* B0  = W0T + ((size_t)lt * 8 + h) * (M_ * D_);
    const float*  b0p = kb.b0[lt] + h * M_;
    const float*  w0u = kb.W0[lt] + ((size_t)h * 257 + 256) * M_;
    int tnB = (wave >> 2) * 4;
    for (int tt = 0; tt < 4; ++tt) {
      int tn = tnB + tt;
      const __bf16* arow = Xs + (tmW * 16 + m16) * 256 + h2 * 8;
      const __bf16* brow = B0 + (size_t)(tn * 16 + m16) * 256 + h2 * 8;
      v8f c = {};
#pragma unroll
      for (int k = 0; k < 256; k += 32)
        c = wmma_bf16(frag_a(arow + k), frag_b(brow + k), c);
      int n = tn * 16 + m16;
      float bn = b0p[n], wu = w0u[n];
#pragma unroll
      for (int r = 0; r < 8; ++r) {
        int m = tmW * 16 + r + 8 * h2;
        float xv = c[r] + bn + Us[m] * wu;
        Ys[m * 128 + n] = (__bf16)fmaxf(xv, 0.f);
      }
    }
  }
  __syncthreads();

  // --- stage 2: (64x128)@(128x128) + b1, relu  (output reuses Xs) ----------
  __bf16* Y2 = Xs;
  {
    const __bf16* B1  = W1T + ((size_t)lt * 8 + h) * (M_ * M_);
    const float*  b1p = kb.b1[lt] + h * M_;
    int tnB = (wave >> 2) * 4;
    for (int tt = 0; tt < 4; ++tt) {
      int tn = tnB + tt;
      const __bf16* arow = Ys + (tmW * 16 + m16) * 128 + h2 * 8;
      const __bf16* brow = B1 + (size_t)(tn * 16 + m16) * 128 + h2 * 8;
      v8f c = {};
#pragma unroll
      for (int k = 0; k < 128; k += 32)
        c = wmma_bf16(frag_a(arow + k), frag_b(brow + k), c);
      int n = tn * 16 + m16;
      float bn = b1p[n];
#pragma unroll
      for (int r = 0; r < 8; ++r) {
        int m = tmW * 16 + r + 8 * h2;
        Y2[m * 128 + n] = (__bf16)fmaxf(c[r] + bn, 0.f);
      }
    }
  }
  __syncthreads();

  // --- stage 3: (64x128)@(128x64) + b2 -> global bf16 ----------------------
  {
    const __bf16* B2  = W2T + ((size_t)lt * 8 + h) * (A_ * M_);
    const float*  b2p = kb.b2[lt] + h * A_;
    __bf16* dst = (t == 0 ? kout : vout) +
                  ((((size_t)l * B_ + b) * H_ + h) * V_ + v0) * A_;
    int tnB = (wave >> 2) * 2;
    for (int tt = 0; tt < 2; ++tt) {
      int tn = tnB + tt;
      const __bf16* arow = Y2 + (tmW * 16 + m16) * 128 + h2 * 8;
      const __bf16* brow = B2 + (size_t)(tn * 16 + m16) * 128 + h2 * 8;
      v8f c = {};
#pragma unroll
      for (int k = 0; k < 128; k += 32)
        c = wmma_bf16(frag_a(arow + k), frag_b(brow + k), c);
      int n = tn * 16 + m16;
      float bn = b2p[n];
#pragma unroll
      for (int r = 0; r < 8; ++r) {
        int m = tmW * 16 + r + 8 * h2;
        dst[(size_t)m * A_ + n] = (__bf16)(c[r] + bn);
      }
    }
  }
}

// ---------------------------------------------------------------------------
// Generic WMMA GEMM: out = act(A(rows x K) @ BT^T + bias [+ resid])
// grid: (rowTiles128, colChunks128), 256 threads (8 waves, 1 tile-row each)
// ---------------------------------------------------------------------------
__global__ __launch_bounds__(256) void wmma_gemm_kernel(
    const __bf16* __restrict__ A, const __bf16* __restrict__ BT,
    const float* __restrict__ bias, int K, int N,
    float* __restrict__ outF, __bf16* __restrict__ outB,
    const float* __restrict__ resid, int doRelu) {
  int tid = threadIdx.x, wave = tid >> 5, lane = tid & 31;
  int m16 = lane & 15, h2 = lane >> 4;
  int row0 = blockIdx.x * 128 + wave * 16;
  int n0 = blockIdx.y * 128;

  const __bf16* arow = A + (size_t)(row0 + m16) * K + h2 * 8;
  __builtin_prefetch(arow, 0, 3);

  for (int tn = 0; tn < 8; ++tn) {
    int nc = n0 + tn * 16;
    const __bf16* brow = BT + (size_t)(nc + m16) * K + h2 * 8;
    v8f c = {};
    for (int k = 0; k < K; k += 32)
      c = wmma_bf16(frag_a(arow + k), frag_b(brow + k), c);
    int n = nc + m16;
    float bn = bias ? bias[n] : 0.f;
#pragma unroll
    for (int r = 0; r < 8; ++r) {
      int m = row0 + r + 8 * h2;
      float x = c[r] + bn;
      if (resid) x += resid[(size_t)m * N + n];
      if (doRelu) x = fmaxf(x, 0.f);
      if (outF) outF[(size_t)m * N + n] = x;
      if (outB) outB[(size_t)m * N + n] = (__bf16)x;
    }
  }
}

// ---------------------------------------------------------------------------
// time-MLP: dt -> 128 -> 128 -> 8, one wave per (p,w), grid 4096 x 256
// ---------------------------------------------------------------------------
__global__ __launch_bounds__(256) void time_mlp_kernel(
    const float* __restrict__ times, const int* __restrict__ history,
    const int* __restrict__ perm,
    const float* __restrict__ w0, const float* __restrict__ b0,
    const float* __restrict__ w1, const float* __restrict__ b1,
    const float* __restrict__ w2, const float* __restrict__ b2,
    float* __restrict__ timeb) {
  __shared__ float H1[8][128];
  __shared__ float H2[8][128];
  int tid = threadIdx.x, wave = tid >> 5, lane = tid & 31;
  int pw = blockIdx.x * 8 + wave;  // 0..32767
  int p = pw >> 7;
  float dt = times[history[pw]] - times[perm[p]];

  for (int i = 0; i < 4; ++i) {
    int k = lane + 32 * i;
    H1[wave][k] = fmaxf(w0[k] * dt + b0[k], 0.f);
  }
  __syncthreads();
  for (int i = 0; i < 4; ++i) {
    int k = lane + 32 * i;
    float s = b1[k];
    for (int j = 0; j < 128; ++j) s += H1[wave][j] * w1[j * 128 + k];
    H2[wave][k] = fmaxf(s, 0.f);
  }
  __syncthreads();
  if (lane < 8) {
    float s = b2[lane];
    for (int j = 0; j < 128; ++j) s += H2[wave][j] * w2[j * 8 + lane];
    timeb[(size_t)pw * 8 + lane] = s;
  }
}

// ---------------------------------------------------------------------------
// Attention (per layer): one block per (b,p); scores -> softmax -> weighted V
// ---------------------------------------------------------------------------
__global__ __launch_bounds__(256) void attn_kernel(
    float* __restrict__ attv,              // [B*P][512]
    const __bf16* __restrict__ keys,       // [b][h][V][64] for this layer
    const __bf16* __restrict__ vals,
    const int* __restrict__ history, const unsigned char* __restrict__ mask,
    const int* __restrict__ role, const float* __restrict__ role_emb,
    const float* __restrict__ timeb) {
  __shared__ float qs[HA_];
  __shared__ float sc[H_ * W_];
  __shared__ int idx[W_];
  int bp = blockIdx.x;
  int b = bp >> 8, p = bp & 255;
  int tid = threadIdx.x;

  qs[tid] = attv[(size_t)bp * HA_ + tid];
  qs[tid + 256] = attv[(size_t)bp * HA_ + 256 + tid];
  if (tid < W_) idx[tid] = history[p * W_ + tid];
  __syncthreads();

  const float scale = 0.125f;  // A^{-1/2}
  for (int pair = tid; pair < H_ * W_; pair += 256) {
    int h = pair >> 7, w = pair & 127;
    const __bf16* kp = keys + (((size_t)b * H_ + h) * V_ + idx[w]) * A_;
    float s = 0.f;
    for (int a = 0; a < A_; ++a) s += qs[h * A_ + a] * (float)kp[a];
    s += role_emb[role[p * W_ + w] * H_ + h] + timeb[(size_t)(p * W_ + w) * H_ + h];
    s *= scale;
    if (mask[p * W_ + w]) s = -3.0e38f;
    sc[h * W_ + w] = s;
  }
  __syncthreads();

  int wave = tid >> 5, lane = tid & 31;
  {
    int h = wave;
    float v[4], m = -3.4e38f;
    for (int i = 0; i < 4; ++i) { v[i] = sc[h * W_ + lane + 32 * i]; m = fmaxf(m, v[i]); }
    for (int off = 16; off; off >>= 1) m = fmaxf(m, __shfl_xor(m, off, 32));
    float sum = 0.f;
    for (int i = 0; i < 4; ++i) { v[i] = __expf(v[i] - m); sum += v[i]; }
    for (int off = 16; off; off >>= 1) sum += __shfl_xor(sum, off, 32);
    float inv = 1.f / sum;
    for (int i = 0; i < 4; ++i) sc[h * W_ + lane + 32 * i] = v[i] * inv;
  }
  __syncthreads();

  for (int pair = tid; pair < H_ * A_; pair += 256) {
    int h = pair >> 6, a = pair & 63;
    const __bf16* vp = vals + ((size_t)b * H_ + h) * V_ * A_ + a;
    float s = 0.f;
    for (int w = 0; w < W_; ++w) s += sc[h * W_ + w] * (float)vp[(size_t)idx[w] * A_];
    attv[(size_t)bp * HA_ + h * A_ + a] += s;
  }
}

// ---------------------------------------------------------------------------
// LayerNorm over 512, in-place, optional bf16 mirror; one block per row
// ---------------------------------------------------------------------------
__global__ __launch_bounds__(256) void ln_kernel(float* __restrict__ x,
                                                 const float* __restrict__ g,
                                                 const float* __restrict__ bt,
                                                 __bf16* __restrict__ xb) {
  __shared__ float red[8];
  int row = blockIdx.x, tid = threadIdx.x;
  int wave = tid >> 5, lane = tid & 31;
  float* xr = x + (size_t)row * HA_;
  float a = xr[tid], b = xr[tid + 256];

  float s = a + b;
  for (int off = 16; off; off >>= 1) s += __shfl_xor(s, off, 32);
  if (lane == 0) red[wave] = s;
  __syncthreads();
  float mean = (red[0] + red[1] + red[2] + red[3] + red[4] + red[5] + red[6] + red[7]) *
               (1.f / 512.f);
  float da = a - mean, db = b - mean;
  float q = da * da + db * db;
  for (int off = 16; off; off >>= 1) q += __shfl_xor(q, off, 32);
  __syncthreads();
  if (lane == 0) red[wave] = q;
  __syncthreads();
  float var = (red[0] + red[1] + red[2] + red[3] + red[4] + red[5] + red[6] + red[7]) *
              (1.f / 512.f);
  float rs = rsqrtf(var + 1e-5f);
  float ya = da * rs * g[tid] + bt[tid];
  float yb = db * rs * g[tid + 256] + bt[tid + 256];
  xr[tid] = ya;
  xr[tid + 256] = yb;
  if (xb) {
    xb[(size_t)row * HA_ + tid] = (__bf16)ya;
    xb[(size_t)row * HA_ + 256 + tid] = (__bf16)yb;
  }
}

// ---------------------------------------------------------------------------
// dist head + NLL: one wave per (b,p), atomicAdd into out[b]
// ---------------------------------------------------------------------------
__global__ __launch_bounds__(256) void dist_kernel(
    const float* __restrict__ attv, const float* __restrict__ W0,
    const float* __restrict__ b0, const float* __restrict__ W1,
    const float* __restrict__ b1, const float* __restrict__ W2,
    const float* __restrict__ b2, const float* __restrict__ trueu,
    const int* __restrict__ perm, float* __restrict__ out) {
  __shared__ float xs[8][HA_];
  __shared__ float h1[8][128];
  __shared__ float h2[8][128];
  __shared__ float lg[8][16];
  int tid = threadIdx.x, wave = tid >> 5, lane = tid & 31;
  int bp = blockIdx.x * 8 + wave;
  int b = bp >> 8, p = bp & 255;

  for (int i = 0; i < 16; ++i) xs[wave][lane + 32 * i] = attv[(size_t)bp * HA_ + lane + 32 * i];
  __syncthreads();
  for (int i = 0; i < 4; ++i) {
    int k = lane + 32 * i;
    float s = b0[k];
    for (int j = 0; j < HA_; ++j) s += xs[wave][j] * W0[j * 128 + k];
    h1[wave][k] = fmaxf(s, 0.f);
  }
  __syncthreads();
  for (int i = 0; i < 4; ++i) {
    int k = lane + 32 * i;
    float s = b1[k];
    for (int j = 0; j < 128; ++j) s += h1[wave][j] * W1[j * 128 + k];
    h2[wave][k] = fmaxf(s, 0.f);
  }
  __syncthreads();
  if (lane < R_) {
    float s = b2[lane];
    for (int j = 0; j < 128; ++j) s += h2[wave][j] * W2[j * R_ + lane];
    lg[wave][lane] = s;
  }
  __syncthreads();
  if (lane == 0) {
    float m = -3.4e38f;
    for (int r = 0; r < R_; ++r) m = fmaxf(m, lg[wave][r]);
    float sum = 0.f;
    for (int r = 0; r < R_; ++r) sum += __expf(lg[wave][r] - m);
    float lse = m + __logf(sum);
    float u = trueu[(size_t)b * V_ + perm[p]];
    int tgt = (int)floorf(u * (float)R_);
    tgt = tgt < 0 ? 0 : (tgt > R_ - 1 ? R_ - 1 : tgt);
    float lp = __logf((float)R_) + lg[wave][tgt] - lse;
    atomicAdd(out + b, -lp);
  }
}

// ---------------------------------------------------------------------------
// Host launcher
// ---------------------------------------------------------------------------
extern "C" void kernel_launch(void* const* d_in, const int* in_sizes, int n_in,
                              void* d_out, int out_size, void* d_ws, size_t ws_size,
                              hipStream_t stream) {
  (void)in_sizes; (void)n_in; (void)out_size; (void)ws_size;
  // ---- input leaf map (setup_inputs insertion order, depth-first) ----------
  const float* encoded = (const float*)d_in[0];
  const float* true_u  = (const float*)d_in[1];
  const float* times   = (const float*)d_in[2];
  const float* dsW = (const float*)d_in[3];
  const float* dsB = (const float*)d_in[4];
  const float *roleEmb[2], *tW0[2], *tB0[2], *tW1[2], *tB1[2], *tW2[2], *tB2[2];
  const float *kW0[2], *kB0[2], *kW1[2], *kB1[2], *kW2[2], *kB2[2];
  const float *vW0[2], *vB0[2], *vW1[2], *vB1[2], *vW2[2], *vB2[2];
  const float *ln1g[2], *ln1b[2], *ffW1[2], *ffB1[2], *ffW2[2], *ffB2[2], *ln2g[2], *ln2b[2];
  for (int l = 0; l < 2; ++l) {
    int base = 5 + l * 27;
    roleEmb[l] = (const float*)d_in[base + 0];
    tW0[l] = (const float*)d_in[base + 1];  tB0[l] = (const float*)d_in[base + 2];
    tW1[l] = (const float*)d_in[base + 3];  tB1[l] = (const float*)d_in[base + 4];
    tW2[l] = (const float*)d_in[base + 5];  tB2[l] = (const float*)d_in[base + 6];
    kW0[l] = (const float*)d_in[base + 7];  kB0[l] = (const float*)d_in[base + 8];
    kW1[l] = (const float*)d_in[base + 9];  kB1[l] = (const float*)d_in[base + 10];
    kW2[l] = (const float*)d_in[base + 11]; kB2[l] = (const float*)d_in[base + 12];
    vW0[l] = (const float*)d_in[base + 13]; vB0[l] = (const float*)d_in[base + 14];
    vW1[l] = (const float*)d_in[base + 15]; vB1[l] = (const float*)d_in[base + 16];
    vW2[l] = (const float*)d_in[base + 17]; vB2[l] = (const float*)d_in[base + 18];
    ln1g[l] = (const float*)d_in[base + 19]; ln1b[l] = (const float*)d_in[base + 20];
    ffW1[l] = (const float*)d_in[base + 21]; ffB1[l] = (const float*)d_in[base + 22];
    ffW2[l] = (const float*)d_in[base + 23]; ffB2[l] = (const float*)d_in[base + 24];
    ln2g[l] = (const float*)d_in[base + 25]; ln2b[l] = (const float*)d_in[base + 26];
  }
  const float* distW0 = (const float*)d_in[59];
  const float* distB0 = (const float*)d_in[60];
  const float* distW1 = (const float*)d_in[61];
  const float* distB1 = (const float*)d_in[62];
  const float* distW2 = (const float*)d_in[63];
  const float* distB2 = (const float*)d_in[64];
  const int* perm = (const int*)d_in[65];
  const int* history = (const int*)d_in[66];
  const unsigned char* mask = (const unsigned char*)d_in[67];
  const int* role = (const int*)d_in[68];

  // ---- workspace carve-up --------------------------------------------------
  char* wsb = (char*)d_ws;
  size_t off = 0;
  auto carve = [&](size_t bytes) -> char* {
    char* r = wsb + off;
    off += (bytes + 255) & ~(size_t)255;
    return r;
  };
  __bf16* WdsT  = (__bf16*)carve((size_t)HA_ * D_ * 2);
  __bf16* ffW1T = (__bf16*)carve((size_t)2 * HA_ * HA_ * 2);
  __bf16* ffW2T = (__bf16*)carve((size_t)2 * HA_ * HA_ * 2);
  __bf16* kvW0T = (__bf16*)carve((size_t)32 * M_ * D_ * 2);
  __bf16* kvW1T = (__bf16*)carve((size_t)32 * M_ * M_ * 2);
  __bf16* kvW2T = (__bf16*)carve((size_t)32 * A_ * M_ * 2);
  __bf16* kvx   = (__bf16*)carve((size_t)B_ * V_ * D_ * 2);
  __bf16* predx = (__bf16*)carve((size_t)B_ * P_ * D_ * 2);
  __bf16* keysB = (__bf16*)carve((size_t)2 * B_ * H_ * V_ * A_ * 2);
  __bf16* valsB = (__bf16*)carve((size_t)2 * B_ * H_ * V_ * A_ * 2);
  float*  attv  = (float*)carve((size_t)B_ * P_ * HA_ * 4);
  __bf16* xn    = (__bf16*)carve((size_t)B_ * P_ * HA_ * 2);
  __bf16* ffh1  = (__bf16*)carve((size_t)B_ * P_ * HA_ * 2);
  float*  timeb = (float*)carve((size_t)P_ * W_ * H_ * 4);

  // ---- prep ----------------------------------------------------------------
  zero_kernel<<<1, 32, 0, stream>>>((float*)d_out, B_);
  f2b_kernel<<<2048, 256, 0, stream>>>(encoded, kvx, (long)B_ * V_ * D_);
  convT_kernel<<<1024, 256, 0, stream>>>(dsW, WdsT, 1, D_, HA_, D_, 0, 0);
  for (int l = 0; l < 2; ++l) {
    convT_kernel<<<1024, 256, 0, stream>>>(ffW1[l], ffW1T + (size_t)l * HA_ * HA_,
                                           1, HA_, HA_, HA_, 0, 0);
    convT_kernel<<<1024, 256, 0, stream>>>(ffW2[l], ffW2T + (size_t)l * HA_ * HA_,
                                           1, HA_, HA_, HA_, 0, 0);
    for (int t = 0; t < 2; ++t) {
      int lt = l * 2 + t;
      const float* w0 = t ? vW0[l] : kW0[l];
      const float* w1 = t ? vW1[l] : kW1[l];
      const float* w2 = t ? vW2[l] : kW2[l];
      convT_kernel<<<1024, 256, 0, stream>>>(w0, kvW0T + (size_t)lt * 8 * M_ * D_,
                                             8, 257, M_, D_, (long)257 * M_, (long)M_ * D_);
      convT_kernel<<<1024, 256, 0, stream>>>(w1, kvW1T + (size_t)lt * 8 * M_ * M_,
                                             8, M_, M_, M_, (long)M_ * M_, (long)M_ * M_);
      convT_kernel<<<1024, 256, 0, stream>>>(w2, kvW2T + (size_t)lt * 8 * A_ * M_,
                                             8, M_, A_, M_, (long)M_ * A_, (long)A_ * M_);
    }
  }
  gather_rows_kernel<<<B_ * P_, 256, 0, stream>>>(kvx, perm, predx);

  // ---- key/value head-MLPs (WMMA) -----------------------------------------
  KvBias kb;
  for (int l = 0; l < 2; ++l) {
    kb.W0[l * 2 + 0] = kW0[l]; kb.W0[l * 2 + 1] = vW0[l];
    kb.b0[l * 2 + 0] = kB0[l]; kb.b0[l * 2 + 1] = vB0[l];
    kb.b1[l * 2 + 0] = kB1[l]; kb.b1[l * 2 + 1] = vB1[l];
    kb.b2[l * 2 + 0] = kB2[l]; kb.b2[l * 2 + 1] = vB2[l];
  }
  kv_mlp_kernel<<<4096, 256, 0, stream>>>(kvx, true_u, kvW0T, kvW1T, kvW2T, kb,
                                          keysB, valsB);

  // ---- att_value init = pred_encoded @ Wds + bds (WMMA) -------------------
  {
    dim3 g(B_ * P_ / 128, HA_ / 128);
    wmma_gemm_kernel<<<g, 256, 0, stream>>>(predx, WdsT, dsB, D_, HA_,
                                            attv, nullptr, nullptr, 0);
  }

  // ---- transformer layers --------------------------------------------------
  for (int l = 0; l < 2; ++l) {
    time_mlp_kernel<<<P_ * W_ / 8, 256, 0, stream>>>(times, history, perm,
                                                     tW0[l], tB0[l], tW1[l], tB1[l],
                                                     tW2[l], tB2[l], timeb);
    attn_kernel<<<B_ * P_, 256, 0, stream>>>(
        attv, keysB + (size_t)l * B_ * H_ * V_ * A_,
        valsB + (size_t)l * B_ * H_ * V_ * A_, history, mask, role, roleEmb[l], timeb);
    ln_kernel<<<B_ * P_, 256, 0, stream>>>(attv, ln1g[l], ln1b[l], xn);
    dim3 g(B_ * P_ / 128, HA_ / 128);
    wmma_gemm_kernel<<<g, 256, 0, stream>>>(xn, ffW1T + (size_t)l * HA_ * HA_,
                                            ffB1[l], HA_, HA_, nullptr, ffh1,
                                            nullptr, 1);
    wmma_gemm_kernel<<<g, 256, 0, stream>>>(ffh1, ffW2T + (size_t)l * HA_ * HA_,
                                            ffB2[l], HA_, HA_, attv, nullptr,
                                            attv, 0);
    ln_kernel<<<B_ * P_, 256, 0, stream>>>(attv, ln2g[l], ln2b[l], nullptr);
  }

  // ---- dist head + loss ----------------------------------------------------
  dist_kernel<<<B_ * P_ / 8, 256, 0, stream>>>(attv, distW0, distB0, distW1, distB1,
                                               distW2, distB2, true_u, perm,
                                               (float*)d_out);
}
